// Pi2Attention_10471130267932
// MI455X (gfx1250) — matrix operand order repaired and checked by reading, not compiled, not run on hardware
//
#include <hip/hip_runtime.h>
#include <hip/hip_bf16.h>
#include <stdint.h>

// ---------------------------------------------------------------------------
// Pi2Attention on MI455X (gfx1250): f16 WMMA (v_wmma_f32_16x16x32_f16) with
// f32 accumulation; tile staging via double-buffered
// global_load_async_to_lds_b128 (ASYNCcnt) so DMA overlaps the WMMAs.
// ---------------------------------------------------------------------------

typedef __attribute__((ext_vector_type(16))) _Float16 v16h;
typedef __attribute__((ext_vector_type(8)))  _Float16 v8h;
typedef __attribute__((ext_vector_type(8)))  float    v8f;

static __device__ __forceinline__ v8f wmma_f16(v16h a, v16h b, v8f c) {
  // D = A(16x32) * B(32x16) + C, f32 accum
  return __builtin_amdgcn_wmma_f32_16x16x32_f16(false, a, false, b,
                                                (short)0, c, false, false);
}

// Async DMA: 16B per lane, global -> LDS, tracked by ASYNCcnt.
// vdst = per-lane LDS byte address (low 32 bits of generic shared pointer),
// vaddr = 64-bit global address.  (ISA 15.18.3, opcode 98)
static __device__ __forceinline__ void async_copy_b128(const _Float16* g,
                                                       _Float16* l) {
  asm volatile("global_load_async_to_lds_b128 %0, %1, off"
               :: "v"((uint32_t)(uintptr_t)l), "v"((uint64_t)(uintptr_t)g)
               : "memory");
}
static __device__ __forceinline__ void wait_async() {
  asm volatile("s_wait_asynccnt 0" ::: "memory");
}

// A-fragment (16x32 f16) from an LDS tile stored row-major [m][k], ld in halves.
// Lanes 0-15 rows M=0..15 hold K {0..7,16..23}; lanes 16-31 hold K {8..15,24..31}.
static __device__ __forceinline__ v16h load_a_frag(const _Float16* t, int ld,
                                                   int m0, int k0) {
  const int lane = threadIdx.x & 31;
  const int r = lane & 15, hi = lane >> 4;
  const _Float16* p = t + (m0 + r) * ld + k0 + hi * 8;
  v8h x0 = *(const v8h*)(p);
  v8h x1 = *(const v8h*)(p + 16);
  v16h o;
#pragma unroll
  for (int i = 0; i < 8; ++i) { o[i] = x0[i]; o[i + 8] = x1[i]; }
  return o;
}

// B-fragment (32x16 f16) from an LDS tile stored transposed [n][k], ld in halves.
static __device__ __forceinline__ v16h load_b_frag(const _Float16* t, int ld,
                                                   int n0, int k0) {
  const int lane = threadIdx.x & 31;
  const int n = lane & 15, hi = lane >> 4;
  const _Float16* p = t + (n0 + n) * ld + k0 + hi * 8;
  v8h x0 = *(const v8h*)(p);
  v8h x1 = *(const v8h*)(p + 16);
  v16h o;
#pragma unroll
  for (int i = 0; i < 8; ++i) { o[i] = x0[i]; o[i + 8] = x1[i]; }
  return o;
}

// ---------------------------------------------------------------------------
// fp32 -> f16 convert with scale (folds pi/2 into weights)
// ---------------------------------------------------------------------------
__global__ void cvt_f32_f16(const float* __restrict__ in,
                            _Float16* __restrict__ out, float scale, int n) {
  int i = blockIdx.x * blockDim.x + threadIdx.x;
  if (i < n) out[i] = (_Float16)(in[i] * scale);
}

// ---------------------------------------------------------------------------
// QKV GEMM: qkv[t][f] = sum_c xh[t][c] * wh[f][c]   (M=4096, N=3072, K=1024)
// Block tile 64x128, BK=32, 256 threads = 8 waves, each wave 32x32 (2x2 frags).
// Double-buffered async-to-LDS staging. Epilogue scatters q (*D^-0.5), k as
// [B,H,T,D] and v transposed [B,H,D,T], all f16.
// ---------------------------------------------------------------------------
__global__ __launch_bounds__(256) void qkv_gemm(
    const _Float16* __restrict__ xh, const _Float16* __restrict__ wh,
    _Float16* __restrict__ qb, _Float16* __restrict__ kb,
    _Float16* __restrict__ vb) {
  __shared__ alignas(16) _Float16 As[2][64 * 40];
  __shared__ alignas(16) _Float16 Bs[2][128 * 40];
  const int tid = threadIdx.x;
  const int wave = tid >> 5;
  const int wm = wave & 1, wn = wave >> 1;
  const int m0 = blockIdx.x * 64;
  const int n0 = blockIdx.y * 128;
  const int rowA = tid >> 2, kcA = (tid & 3) * 8;

  auto stage = [&](int kt, int buf) {
    async_copy_b128(xh + (size_t)(m0 + rowA) * 1024 + kt + kcA,
                    &As[buf][rowA * 40 + kcA]);
#pragma unroll
    for (int i = 0; i < 2; ++i) {
      int c = tid + i * 256;
      int row = c >> 2, kc = (c & 3) * 8;
      async_copy_b128(wh + (size_t)(n0 + row) * 1024 + kt + kc,
                      &Bs[buf][row * 40 + kc]);
    }
  };

  v8f acc[2][2] = {};
  stage(0, 0);
  for (int it = 0; it < 32; ++it) {
    wait_async();
    __syncthreads();
    if (it + 1 < 32) stage((it + 1) * 32, (it + 1) & 1);
    const _Float16* A = As[it & 1];
    const _Float16* B = Bs[it & 1];
    v16h a0 = load_a_frag(A, 40, wm * 32 + 0, 0);
    v16h a1 = load_a_frag(A, 40, wm * 32 + 16, 0);
    v16h b0 = load_b_frag(B, 40, wn * 32 + 0, 0);
    v16h b1 = load_b_frag(B, 40, wn * 32 + 16, 0);
    acc[0][0] = wmma_f16(a0, b0, acc[0][0]);
    acc[0][1] = wmma_f16(a0, b1, acc[0][1]);
    acc[1][0] = wmma_f16(a1, b0, acc[1][0]);
    acc[1][1] = wmma_f16(a1, b1, acc[1][1]);
  }

  const int lane = tid & 31;
  const float scale_attn = 0.125f;  // 64^-0.5
#pragma unroll
  for (int fm = 0; fm < 2; ++fm) {
#pragma unroll
    for (int fn = 0; fn < 2; ++fn) {
      int n = n0 + wn * 32 + fn * 16 + (lane & 15);
      int s = n >> 10;          // 0=q 1=k 2=v
      int hh = (n >> 6) & 15;   // head
      int d = n & 63;           // head dim
#pragma unroll
      for (int e = 0; e < 8; ++e) {
        int m = m0 + wm * 32 + fm * 16 + e + ((lane >> 4) << 3);
        int bb = m >> 11, ti = m & 2047;
        float val = acc[fm][fn][e];
        size_t hbase = (size_t)(bb * 16 + hh);
        if (s == 0)
          qb[(hbase * 2048 + ti) * 64 + d] = (_Float16)(val * scale_attn);
        else if (s == 1)
          kb[(hbase * 2048 + ti) * 64 + d] = (_Float16)val;
        else
          vb[(hbase * 64 + d) * 2048 + ti] = (_Float16)val;  // transposed
      }
    }
  }
}

// ---------------------------------------------------------------------------
// Flash attention: per (b,h,64-query tile). 128 threads = 4 waves, each wave
// owns a 16-row query stripe (Q frags live in registers). K/V tiles are
// double-buffered async-to-LDS. S = Q K^T (WMMA), online softmax via
// shfl_xor row reductions within lane halves, O += P V (WMMA via LDS-staged P).
// ---------------------------------------------------------------------------
__global__ __launch_bounds__(128) void attn_kernel(
    const _Float16* __restrict__ qb, const _Float16* __restrict__ kb,
    const _Float16* __restrict__ vb, _Float16* __restrict__ ob) {
  __shared__ alignas(16) _Float16 Qs[64 * 72];
  __shared__ alignas(16) _Float16 Ks[2][64 * 72];
  __shared__ alignas(16) _Float16 Vs[2][64 * 72];   // [d][j]
  __shared__ alignas(16) _Float16 Ps[4][16 * 72];

  const int tid = threadIdx.x, lane = tid & 31, wave = tid >> 5;
  const int qt = blockIdx.x, hh = blockIdx.y, bb = blockIdx.z;
  const size_t headoff = (size_t)(bb * 16 + hh) * 2048 * 64;
  const _Float16* qhead = qb + headoff;  // [t][d]
  const _Float16* khead = kb + headoff;  // [t][d]
  const _Float16* vhead = vb + headoff;  // [d][t]

  auto stage_kv = [&](int jt, int buf) {
#pragma unroll
    for (int i = 0; i < 4; ++i) {
      int c = tid + i * 128;
      int row = c >> 3, kc = (c & 7) * 8;
      async_copy_b128(khead + (size_t)(jt + row) * 64 + kc,
                      &Ks[buf][row * 72 + kc]);
      async_copy_b128(vhead + (size_t)row * 2048 + jt + kc,
                      &Vs[buf][row * 72 + kc]);
    }
  };

#pragma unroll
  for (int i = 0; i < 4; ++i) {  // Q tile: 64x64 halves = 512 chunks
    int c = tid + i * 128;
    int row = c >> 3, kc = (c & 7) * 8;
    *(v8h*)(Qs + row * 72 + kc) =
        *(const v8h*)(qhead + (size_t)(qt * 64 + row) * 64 + kc);
  }
  stage_kv(0, 0);
  __syncthreads();
  const v16h qa0 = load_a_frag(Qs, 72, wave * 16, 0);
  const v16h qa1 = load_a_frag(Qs, 72, wave * 16, 32);

  float mrow[8], lrow[8];
  v8f o[4] = {};
#pragma unroll
  for (int e = 0; e < 8; ++e) { mrow[e] = -1e30f; lrow[e] = 0.0f; }

  for (int it = 0; it < 32; ++it) {
    wait_async();
    __syncthreads();
    if (it + 1 < 32) stage_kv((it + 1) * 64, (it + 1) & 1);
    const _Float16* K = Ks[it & 1];
    const _Float16* V = Vs[it & 1];

    v8f sf[4];
#pragma unroll
    for (int fn = 0; fn < 4; ++fn) {  // S stripe 16x64
      v8f s = {};
      s = wmma_f16(qa0, load_b_frag(K, 72, fn * 16, 0), s);
      s = wmma_f16(qa1, load_b_frag(K, 72, fn * 16, 32), s);
      sf[fn] = s;
    }

    float corr[8];
#pragma unroll
    for (int e = 0; e < 8; ++e) {  // row max over 64 keys
      float mx = sf[0][e];
      mx = fmaxf(mx, sf[1][e]);
      mx = fmaxf(mx, sf[2][e]);
      mx = fmaxf(mx, sf[3][e]);
      mx = fmaxf(mx, __shfl_xor(mx, 1, 32));
      mx = fmaxf(mx, __shfl_xor(mx, 2, 32));
      mx = fmaxf(mx, __shfl_xor(mx, 4, 32));
      mx = fmaxf(mx, __shfl_xor(mx, 8, 32));
      float mnew = fmaxf(mrow[e], mx);
      corr[e] = __expf(mrow[e] - mnew);
      mrow[e] = mnew;
      lrow[e] *= corr[e];
    }
#pragma unroll
    for (int fd = 0; fd < 4; ++fd)
#pragma unroll
      for (int e = 0; e < 8; ++e) o[fd][e] *= corr[e];

#pragma unroll
    for (int fn = 0; fn < 4; ++fn)
#pragma unroll
      for (int e = 0; e < 8; ++e) sf[fn][e] = __expf(sf[fn][e] - mrow[e]);

#pragma unroll
    for (int e = 0; e < 8; ++e) {  // row sum
      float rs = sf[0][e] + sf[1][e] + sf[2][e] + sf[3][e];
      rs += __shfl_xor(rs, 1, 32);
      rs += __shfl_xor(rs, 2, 32);
      rs += __shfl_xor(rs, 4, 32);
      rs += __shfl_xor(rs, 8, 32);
      lrow[e] += rs;
    }

    _Float16* pp = Ps[wave];  // stage P (C layout -> row-major [i][j])
#pragma unroll
    for (int fn = 0; fn < 4; ++fn)
#pragma unroll
      for (int e = 0; e < 8; ++e)
        pp[(e + ((lane >> 4) << 3)) * 72 + fn * 16 + (lane & 15)] =
            (_Float16)sf[fn][e];
    __syncthreads();

#pragma unroll
    for (int fd = 0; fd < 4; ++fd) {  // O += P V
      o[fd] = wmma_f16(load_a_frag(pp, 72, 0, 0),
                       load_b_frag(V, 72, fd * 16, 0), o[fd]);
      o[fd] = wmma_f16(load_a_frag(pp, 72, 0, 32),
                       load_b_frag(V, 72, fd * 16, 32), o[fd]);
    }
  }

#pragma unroll
  for (int fd = 0; fd < 4; ++fd) {
#pragma unroll
    for (int e = 0; e < 8; ++e) {
      int m = wave * 16 + e + ((lane >> 4) << 3);
      int ti = qt * 64 + m;
      int cc = hh * 64 + fd * 16 + (lane & 15);
      float val = o[fd][e] / lrow[e];
      ob[(size_t)(bb * 2048 + ti) * 1024 + cc] = (_Float16)val;
    }
  }
}

// ---------------------------------------------------------------------------
// Projection GEMM: out[t][f] = sum_c oh[t][c] * wh[f][c]  (4096x1024x1024)
// Same tiling as qkv_gemm; fp32 output.
// ---------------------------------------------------------------------------
__global__ __launch_bounds__(256) void proj_gemm(
    const _Float16* __restrict__ oh, const _Float16* __restrict__ wh,
    float* __restrict__ out) {
  __shared__ alignas(16) _Float16 As[2][64 * 40];
  __shared__ alignas(16) _Float16 Bs[2][128 * 40];
  const int tid = threadIdx.x;
  const int wave = tid >> 5;
  const int wm = wave & 1, wn = wave >> 1;
  const int m0 = blockIdx.x * 64;
  const int n0 = blockIdx.y * 128;
  const int rowA = tid >> 2, kcA = (tid & 3) * 8;

  auto stage = [&](int kt, int buf) {
    async_copy_b128(oh + (size_t)(m0 + rowA) * 1024 + kt + kcA,
                    &As[buf][rowA * 40 + kcA]);
#pragma unroll
    for (int i = 0; i < 2; ++i) {
      int c = tid + i * 256;
      int row = c >> 2, kc = (c & 3) * 8;
      async_copy_b128(wh + (size_t)(n0 + row) * 1024 + kt + kc,
                      &Bs[buf][row * 40 + kc]);
    }
  };

  v8f acc[2][2] = {};
  stage(0, 0);
  for (int it = 0; it < 32; ++it) {
    wait_async();
    __syncthreads();
    if (it + 1 < 32) stage((it + 1) * 32, (it + 1) & 1);
    const _Float16* A = As[it & 1];
    const _Float16* B = Bs[it & 1];
    v16h a0 = load_a_frag(A, 40, wm * 32 + 0, 0);
    v16h a1 = load_a_frag(A, 40, wm * 32 + 16, 0);
    v16h b0 = load_b_frag(B, 40, wn * 32 + 0, 0);
    v16h b1 = load_b_frag(B, 40, wn * 32 + 16, 0);
    acc[0][0] = wmma_f16(a0, b0, acc[0][0]);
    acc[0][1] = wmma_f16(a0, b1, acc[0][1]);
    acc[1][0] = wmma_f16(a1, b0, acc[1][0]);
    acc[1][1] = wmma_f16(a1, b1, acc[1][1]);
  }

  const int lane = tid & 31;
#pragma unroll
  for (int fm = 0; fm < 2; ++fm) {
#pragma unroll
    for (int fn = 0; fn < 2; ++fn) {
      int n = n0 + wn * 32 + fn * 16 + (lane & 15);
#pragma unroll
      for (int e = 0; e < 8; ++e) {
        int m = m0 + wm * 32 + fm * 16 + e + ((lane >> 4) << 3);
        out[(size_t)m * 1024 + n] = acc[fm][fn][e];
      }
    }
  }
}

// ---------------------------------------------------------------------------
extern "C" void kernel_launch(void* const* d_in, const int* in_sizes, int n_in,
                              void* d_out, int out_size, void* d_ws,
                              size_t ws_size, hipStream_t stream) {
  (void)in_sizes; (void)n_in; (void)out_size; (void)ws_size;
  const float* x     = (const float*)d_in[0];  // [2,2048,1024]
  const float* wqkv  = (const float*)d_in[1];  // [3072,1024]
  const float* wproj = (const float*)d_in[2];  // [1024,1024]
  float* out = (float*)d_out;                  // [2,2048,1024]

  char* ws = (char*)d_ws;
  _Float16* xh     = (_Float16*)(ws);                       //  8 MB
  _Float16* wqkvh  = (_Float16*)(ws + 8388608ull);          //  6 MB
  _Float16* wprojh = (_Float16*)(ws + 14680064ull);         //  2 MB
  _Float16* qbuf   = (_Float16*)(ws + 16777216ull);         //  8 MB [B,H,T,D]
  _Float16* kbuf   = (_Float16*)(ws + 25165824ull);         //  8 MB [B,H,T,D]
  _Float16* vbuf   = (_Float16*)(ws + 33554432ull);         //  8 MB [B,H,D,T]
  _Float16* obuf   = (_Float16*)(ws + 41943040ull);         //  8 MB [B,T,C]

  const float HPI = 1.57079632679489662f;
  cvt_f32_f16<<<(4194304 + 255) / 256, 256, 0, stream>>>(x, xh, 1.0f, 4194304);
  cvt_f32_f16<<<(3145728 + 255) / 256, 256, 0, stream>>>(wqkv, wqkvh, HPI, 3145728);
  cvt_f32_f16<<<(1048576 + 255) / 256, 256, 0, stream>>>(wproj, wprojh, HPI, 1048576);

  qkv_gemm<<<dim3(64, 24), 256, 0, stream>>>(xh, wqkvh, qbuf, kbuf, vbuf);
  attn_kernel<<<dim3(32, 16, 2), 128, 0, stream>>>(qbuf, kbuf, vbuf, obuf);
  proj_gemm<<<dim3(64, 8), 256, 0, stream>>>(obuf, wprojh, out);
}
